// MultiHeadSelfAttention_33320356282661
// MI455X (gfx1250) — matrix operand order repaired and checked by reading, not compiled, not run on hardware
//
#include <hip/hip_runtime.h>
#include <hip/hip_bf16.h>

#define L_SEQ  4096
#define F_IN_  1024
#define NHEAD  16
#define DHEAD  64
#define DMODEL 1024   // NHEAD * DHEAD
#define F_OUT_ 1024

// padded LDS strides (elements) -> conflict-free ds_load_b128 fragment gathers
#define LAP  40   // gemm A tile row stride   (128 x 32 tile)
#define LBP  40   // gemm B^T tile row stride (128 x 32 tile)
#define SKP  72   // attn K tile row stride   (128 x 64 tile)
#define SVP  136  // attn V^T tile row stride (64 x 128 tile)

typedef __attribute__((ext_vector_type(16))) __bf16 v16bf;
typedef __attribute__((ext_vector_type(8)))  float  v8f;

union Frag16 { v16bf v; uint4 q[2]; unsigned short u[16]; };
union Pack8  { uint4 q; unsigned short u[8]; };

__device__ __forceinline__ unsigned short f2bf(float f) {
  union { float f; unsigned u; } c; c.f = f;
  unsigned r = c.u + 0x7FFFu + ((c.u >> 16) & 1u);   // round-to-nearest-even
  return (unsigned short)(r >> 16);
}
__device__ __forceinline__ unsigned pack2(unsigned short lo, unsigned short hi) {
  return (unsigned)lo | ((unsigned)hi << 16);
}

// ---- CDNA5 async global->LDS copy (ASYNCcnt-tracked, no VGPR data) ---------
// global_load_async_to_lds_b128 vdst, vaddr, off
//   vdst  = 32-bit LDS byte address (low 32 bits of the generic LDS pointer)
//   vaddr = 64-bit global address (VGPR pair)
__device__ __forceinline__ void async_copy_b128(const void* g, void* l) {
  unsigned lds_addr = (unsigned)(size_t)l;
  asm volatile("global_load_async_to_lds_b128 %0, %1, off"
               :: "v"(lds_addr), "v"(g)
               : "memory");
}
__device__ __forceinline__ void wait_async0() {
  asm volatile("s_wait_asynccnt 0x0" ::: "memory");
}

// ---------------------------------------------------------------- convert ---
__global__ void cvt_f32_bf16(const float* __restrict__ src,
                             unsigned short* __restrict__ dst, int n4,
                             float scale) {
  int i = blockIdx.x * blockDim.x + threadIdx.x;
  if (i < n4) {
    float4 v = ((const float4*)src)[i];
    uint2 o;
    o.x = pack2(f2bf(v.x * scale), f2bf(v.y * scale));
    o.y = pack2(f2bf(v.z * scale), f2bf(v.w * scale));
    ((uint2*)dst)[i] = o;
  }
}

// ------------------------------------------------------------------- GEMM ---
// C[M,N] = A[M,K](bf16) * B[K,N](bf16) + bias[N]*bscale; 128x128 block tile,
// 8 waves, wave = 32x64 via 2x4 WMMA 16x16x32 bf16. A tile streamed with
// async global->LDS (double buffered); B^T tile register-repacked (transpose).
template <bool OUT_BF16>
__global__ __launch_bounds__(256) void gemm_bf16_128x128(
    const unsigned short* __restrict__ A,
    const unsigned short* __restrict__ B,
    const float* __restrict__ bias, float bscale,
    void* __restrict__ Cout, int M, int N, int K) {
  __shared__ unsigned short lA[2][128 * LAP];
  __shared__ unsigned short lBt[2][128 * LBP];

  const int tid   = threadIdx.x;
  const int lane  = tid & 31;
  const int wave  = tid >> 5;
  const int wm    = (wave >> 1) * 32;
  const int wn    = (wave & 1) * 64;
  const int m0    = blockIdx.y * 128;
  const int n0    = blockIdx.x * 128;
  const int lhalf = lane >> 4;
  const int l15   = lane & 15;

  const int arow = tid >> 1, aseg = (tid & 1) * 16;
  const int bkp  = tid >> 4, bcol = (tid & 15) * 8;

  v8f acc[2][4];
#pragma unroll
  for (int i = 0; i < 2; ++i)
#pragma unroll
    for (int j = 0; j < 4; ++j) acc[i][j] = {};

  Pack8 sb0, sb1;
  {   // prologue: async A tile 0, B regs tile 0
    const unsigned short* ga = A + (size_t)(m0 + arow) * K + aseg;
    async_copy_b128(ga,     &lA[0][arow * LAP + aseg]);
    async_copy_b128(ga + 8, &lA[0][arow * LAP + aseg + 8]);
    sb0.q = *(const uint4*)(B + (size_t)(2 * bkp)     * N + n0 + bcol);
    sb1.q = *(const uint4*)(B + (size_t)(2 * bkp + 1) * N + n0 + bcol);
  }

  for (int k0 = 0; k0 < K; k0 += 32) {
    const int buf = (k0 >> 5) & 1;
    {   // B^T regs -> LDS (k-pairs packed as b32)
#pragma unroll
      for (int e = 0; e < 8; ++e)
        *(unsigned*)(&lBt[buf][(bcol + e) * LBP + 2 * bkp]) = pack2(sb0.u[e], sb1.u[e]);
    }
    wait_async0();        // this wave's async A tile for `buf` has landed
    __syncthreads();      // -> visible to all waves
    if (k0 + 32 < K) {    // stream next tile while WMMAs run
      const int kn = k0 + 32;
      const unsigned short* ga = A + (size_t)(m0 + arow) * K + kn + aseg;
      async_copy_b128(ga,     &lA[buf ^ 1][arow * LAP + aseg]);
      async_copy_b128(ga + 8, &lA[buf ^ 1][arow * LAP + aseg + 8]);
      sb0.q = *(const uint4*)(B + (size_t)(kn + 2 * bkp)     * N + n0 + bcol);
      sb1.q = *(const uint4*)(B + (size_t)(kn + 2 * bkp + 1) * N + n0 + bcol);
    }

    Frag16 af[2], bfrg[4];
    const int kbA = lhalf ? 8 : 0;
#pragma unroll
    for (int mf = 0; mf < 2; ++mf) {
      int row = wm + mf * 16 + l15;
      af[mf].q[0] = *(const uint4*)(&lA[buf][row * LAP + kbA]);
      af[mf].q[1] = *(const uint4*)(&lA[buf][row * LAP + kbA + 16]);
    }
    const int kbB = lhalf ? 8 : 0;
#pragma unroll
    for (int nf = 0; nf < 4; ++nf) {
      int col = wn + nf * 16 + l15;
      bfrg[nf].q[0] = *(const uint4*)(&lBt[buf][col * LBP + kbB]);
      bfrg[nf].q[1] = *(const uint4*)(&lBt[buf][col * LBP + kbB + 16]);
    }
#pragma unroll
    for (int mf = 0; mf < 2; ++mf)
#pragma unroll
      for (int nf = 0; nf < 4; ++nf)
        acc[mf][nf] = __builtin_amdgcn_wmma_f32_16x16x32_bf16(
            false, af[mf].v, false, bfrg[nf].v, (short)0, acc[mf][nf],
            false, false);
  }

#pragma unroll
  for (int mf = 0; mf < 2; ++mf)
#pragma unroll
    for (int nf = 0; nf < 4; ++nf) {
      int col  = n0 + wn + nf * 16 + l15;
      float b  = bias ? bias[col] * bscale : 0.0f;
      int rowb = m0 + wm + mf * 16 + lhalf * 8;
#pragma unroll
      for (int r = 0; r < 8; ++r) {
        float v = acc[mf][nf][r] + b;
        if (OUT_BF16)
          ((unsigned short*)Cout)[(size_t)(rowb + r) * N + col] = f2bf(v);
        else
          ((float*)Cout)[(size_t)(rowb + r) * N + col] = v;
      }
    }
}

// -------------------------------------------------- fused flash attention ---
// grid = (L/128, H). 8 waves, wave = 16 query rows. KV chunks of 128 keys,
// double-buffered: K tile via async global->LDS, V^T register-repacked.
// Q carries the 1/sqrt(dk) scale (folded into Wq/bq), so S comes out scaled.
__global__ __launch_bounds__(256) void attn_fused(
    const unsigned short* __restrict__ Qb,
    const unsigned short* __restrict__ Kb,
    const unsigned short* __restrict__ Vb,
    unsigned short* __restrict__ vals) {
  __shared__ unsigned short sK[2][128 * SKP];   // [key][dh]
  __shared__ unsigned short sVt[2][64 * SVP];   // [dh][key]
  __shared__ unsigned short sP[8][16 * 32];     // per-wave P tile [row][key]

  const int tid   = threadIdx.x;
  const int lane  = tid & 31;
  const int wave  = tid >> 5;
  const int lhalf = lane >> 4;
  const int l15   = lane & 15;
  const int h     = blockIdx.y;
  const int hc    = h * DHEAD;
  const int q0    = blockIdx.x * 128 + wave * 16;

  const int krow = tid >> 1, kseg = (tid & 1) * 32;   // K: 4 async b128
  const int vkp  = tid >> 2, vseg = (tid & 3) * 16;   // V: 2 rows x 16 dh

  Frag16 qf[2];
  {
    int row = q0 + l15;
    int kb  = lhalf ? 8 : 0;
    const unsigned short* base = Qb + (size_t)row * DMODEL + hc;
#pragma unroll
    for (int f = 0; f < 2; ++f) {
      qf[f].q[0] = *(const uint4*)(base + f * 32 + kb);
      qf[f].q[1] = *(const uint4*)(base + f * 32 + kb + 16);
    }
  }

  v8f o[4];
#pragma unroll
  for (int f = 0; f < 4; ++f) o[f] = {};
  float mrow[8], lrow[8];
#pragma unroll
  for (int r = 0; r < 8; ++r) { mrow[r] = -1e30f; lrow[r] = 0.0f; }

  struct VStage { Pack8 lo[2], hi[2]; };
  VStage vs;
  {   // prologue: async K chunk 0, V regs chunk 0
    const unsigned short* gk = Kb + (size_t)krow * DMODEL + hc + kseg;
#pragma unroll
    for (int s = 0; s < 4; ++s)
      async_copy_b128(gk + s * 8, &sK[0][krow * SKP + kseg + s * 8]);
    const uint4* glo = (const uint4*)(Vb + (size_t)(2 * vkp)     * DMODEL + hc + vseg);
    const uint4* ghi = (const uint4*)(Vb + (size_t)(2 * vkp + 1) * DMODEL + hc + vseg);
    vs.lo[0].q = glo[0]; vs.lo[1].q = glo[1];
    vs.hi[0].q = ghi[0]; vs.hi[1].q = ghi[1];
  }

  for (int kv0 = 0; kv0 < L_SEQ; kv0 += 128) {
    const int buf = (kv0 >> 7) & 1;
    {   // V^T regs -> LDS (key pairs packed as b32)
#pragma unroll
      for (int e = 0; e < 16; ++e)
        *(unsigned*)(&sVt[buf][(vseg + e) * SVP + 2 * vkp]) =
            pack2(vs.lo[e >> 3].u[e & 7], vs.hi[e >> 3].u[e & 7]);
    }
    wait_async0();        // this wave's async K tile for `buf` has landed
    __syncthreads();
    if (kv0 + 128 < L_SEQ) {   // stream next chunk during compute
      const int kn = kv0 + 128;
      const unsigned short* gk = Kb + (size_t)(kn + krow) * DMODEL + hc + kseg;
#pragma unroll
      for (int s = 0; s < 4; ++s)
        async_copy_b128(gk + s * 8, &sK[buf ^ 1][krow * SKP + kseg + s * 8]);
      const uint4* glo = (const uint4*)(Vb + (size_t)(kn + 2 * vkp)     * DMODEL + hc + vseg);
      const uint4* ghi = (const uint4*)(Vb + (size_t)(kn + 2 * vkp + 1) * DMODEL + hc + vseg);
      vs.lo[0].q = glo[0]; vs.lo[1].q = glo[1];
      vs.hi[0].q = ghi[0]; vs.hi[1].q = ghi[1];
    }

#pragma unroll
    for (int kt = 0; kt < 128; kt += 32) {
      const int kbB = lhalf ? 8 : 0;
      Frag16 kf[2][2];
#pragma unroll
      for (int j = 0; j < 2; ++j) {
        int key = kt + j * 16 + l15;
#pragma unroll
        for (int f = 0; f < 2; ++f) {
          kf[j][f].q[0] = *(const uint4*)(&sK[buf][key * SKP + f * 32 + kbB]);
          kf[j][f].q[1] = *(const uint4*)(&sK[buf][key * SKP + f * 32 + kbB + 16]);
        }
      }
      v8f s01[2];
#pragma unroll
      for (int j = 0; j < 2; ++j) {
        s01[j] = {};
        s01[j] = __builtin_amdgcn_wmma_f32_16x16x32_bf16(
            false, qf[0].v, false, kf[j][0].v, (short)0, s01[j], false, false);
        s01[j] = __builtin_amdgcn_wmma_f32_16x16x32_bf16(
            false, qf[1].v, false, kf[j][1].v, (short)0, s01[j], false, false);
      }
      Frag16 vf[4];
#pragma unroll
      for (int f = 0; f < 4; ++f) {
        int dh = f * 16 + l15;
        vf[f].q[0] = *(const uint4*)(&sVt[buf][dh * SVP + kt + kbB]);
        vf[f].q[1] = *(const uint4*)(&sVt[buf][dh * SVP + kt + kbB + 16]);
      }
      float p0[8], p1[8];
#pragma unroll
      for (int r = 0; r < 8; ++r) {
        float a = s01[0][r];           // already scaled by 1/sqrt(dk)
        float b = s01[1][r];
        float mx = fmaxf(a, b);
        mx = fmaxf(mx, __shfl_xor(mx, 1, 32));
        mx = fmaxf(mx, __shfl_xor(mx, 2, 32));
        mx = fmaxf(mx, __shfl_xor(mx, 4, 32));
        mx = fmaxf(mx, __shfl_xor(mx, 8, 32));
        float mnew = fmaxf(mrow[r], mx);
        float corr = __expf(mrow[r] - mnew);
        mrow[r] = mnew;
        float ea = __expf(a - mnew);
        float eb = __expf(b - mnew);
        p0[r] = ea; p1[r] = eb;
        lrow[r] = lrow[r] * corr + (ea + eb);   // per-lane partial sum
#pragma unroll
        for (int f = 0; f < 4; ++f) o[f][r] *= corr;
      }
      unsigned short* pw = sP[wave];
#pragma unroll
      for (int r = 0; r < 8; ++r) {
        int row = r + lhalf * 8;
        pw[row * 32 + l15]      = f2bf(p0[r]);
        pw[row * 32 + 16 + l15] = f2bf(p1[r]);
      }
      Frag16 pf;
      {
        int row = l15, kb = lhalf ? 8 : 0;
        pf.q[0] = *(const uint4*)(&pw[row * 32 + kb]);
        pf.q[1] = *(const uint4*)(&pw[row * 32 + kb + 16]);
      }
#pragma unroll
      for (int f = 0; f < 4; ++f)
        o[f] = __builtin_amdgcn_wmma_f32_16x16x32_bf16(
            false, pf.v, false, vf[f].v, (short)0, o[f], false, false);
    }
  }

#pragma unroll
  for (int r = 0; r < 8; ++r) {
    float s = lrow[r];
    s += __shfl_xor(s, 1, 32);
    s += __shfl_xor(s, 2, 32);
    s += __shfl_xor(s, 4, 32);
    s += __shfl_xor(s, 8, 32);
    lrow[r] = 1.0f / s;
  }
#pragma unroll
  for (int f = 0; f < 4; ++f) {
    int col = hc + f * 16 + l15;
#pragma unroll
    for (int r = 0; r < 8; ++r) {
      int row = q0 + r + lhalf * 8;
      vals[(size_t)row * DMODEL + col] = f2bf(o[f][r] * lrow[r]);
    }
  }
}

// ----------------------------------------------------------------- launch ---
extern "C" void kernel_launch(void* const* d_in, const int* in_sizes, int n_in,
                              void* d_out, int out_size, void* d_ws, size_t ws_size,
                              hipStream_t stream) {
  const float* x  = (const float*)d_in[0];
  const float* Wq = (const float*)d_in[1];
  const float* bq = (const float*)d_in[2];
  const float* Wk = (const float*)d_in[3];
  const float* bk = (const float*)d_in[4];
  const float* Wv = (const float*)d_in[5];
  const float* bv = (const float*)d_in[6];
  const float* Wo = (const float*)d_in[7];
  const float* bo = (const float*)d_in[8];
  float* out = (float*)d_out;

  char* ws = (char*)d_ws;
  const size_t MB = 1024u * 1024u;
  unsigned short* xb  = (unsigned short*)(ws + 0 * MB);
  unsigned short* wqb = (unsigned short*)(ws + 8 * MB);
  unsigned short* wkb = (unsigned short*)(ws + 10 * MB);
  unsigned short* wvb = (unsigned short*)(ws + 12 * MB);
  unsigned short* wob = (unsigned short*)(ws + 14 * MB);
  unsigned short* Qb  = (unsigned short*)(ws + 16 * MB);
  unsigned short* Kb  = (unsigned short*)(ws + 24 * MB);
  unsigned short* Vb  = (unsigned short*)(ws + 32 * MB);
  unsigned short* Vl  = (unsigned short*)(ws + 40 * MB);

  const float qscale = 0.125f;   // 1/sqrt(DHEAD), folded into Wq/bq
  const int nx4 = L_SEQ * F_IN_ / 4;
  const int nw4 = F_IN_ * DMODEL / 4;
  cvt_f32_bf16<<<(nx4 + 255) / 256, 256, 0, stream>>>(x, xb, nx4, 1.0f);
  cvt_f32_bf16<<<(nw4 + 255) / 256, 256, 0, stream>>>(Wq, wqb, nw4, qscale);
  cvt_f32_bf16<<<(nw4 + 255) / 256, 256, 0, stream>>>(Wk, wkb, nw4, 1.0f);
  cvt_f32_bf16<<<(nw4 + 255) / 256, 256, 0, stream>>>(Wv, wvb, nw4, 1.0f);
  cvt_f32_bf16<<<(nw4 + 255) / 256, 256, 0, stream>>>(Wo, wob, nw4, 1.0f);

  dim3 g(DMODEL / 128, L_SEQ / 128);
  gemm_bf16_128x128<true><<<g, 256, 0, stream>>>(xb, wqb, bq, qscale, Qb, L_SEQ, DMODEL, F_IN_);
  gemm_bf16_128x128<true><<<g, 256, 0, stream>>>(xb, wkb, bk, 1.0f, Kb, L_SEQ, DMODEL, F_IN_);
  gemm_bf16_128x128<true><<<g, 256, 0, stream>>>(xb, wvb, bv, 1.0f, Vb, L_SEQ, DMODEL, F_IN_);

  attn_fused<<<dim3(L_SEQ / 128, NHEAD), 256, 0, stream>>>(Qb, Kb, Vb, Vl);

  gemm_bf16_128x128<false><<<dim3(F_OUT_ / 128, L_SEQ / 128), 256, 0, stream>>>(
      Vl, wob, bo, 1.0f, out, L_SEQ, F_OUT_, DMODEL);
}
